// QualityNetAttn_55387898249506
// MI455X (gfx1250) — compile-verified, matrix operand order checked by reading, not run on hardware
//
#include <hip/hip_runtime.h>
#include <hip/hip_bf16.h>

typedef __attribute__((ext_vector_type(16))) _Float16 v16h;
typedef __attribute__((ext_vector_type(8)))  float    v8f;

#define LA 2048   // sequence length
#define BB 8      // batch
#define HLS 100   // hidden per dir

// ---- CDNA5 async global->LDS copy via inline asm (ISA 08_async_tensor §4) --
#if defined(__has_builtin)
#if __has_builtin(__builtin_amdgcn_s_wait_asynccnt)
#define HAVE_WAIT_ASYNC_BUILTIN 1
#endif
#endif

__device__ __forceinline__ void wait_asynccnt0() {
#if defined(HAVE_WAIT_ASYNC_BUILTIN)
    __builtin_amdgcn_s_wait_asynccnt(0);
#else
    asm volatile("s_wait_asynccnt 0x0" ::: "memory");
#endif
}

// 16B async copy: VDST = LDS byte address (low 32 bits of generic pointer),
// VADDR = 64-bit global address, SADDR = off. Tracked by ASYNCcnt.
__device__ __forceinline__ void async_copy_b128(void* lds_dst, const void* gsrc) {
    unsigned int lds_addr = (unsigned int)(unsigned long long)lds_dst;
    asm volatile("global_load_async_to_lds_b128 %0, %1, off"
                 :
                 : "v"(lds_addr), "v"(gsrc)
                 : "memory");
}

// ---------------------------------------------------------------------------
// WMMA fragment loaders (CDNA5 16x16x32 f16 layouts, wave32)
// A (16xK tile, row-major): lanes 0-15 = rows, lane half selects K sub-block.
// VGPR v holds K pair at base = (v&4?16:0) + half*8 + (v&3)*2.
// ---------------------------------------------------------------------------
__device__ __forceinline__ v16h load_a16(const _Float16* A, int lda, int lane) {
    int m = lane & 15, half = lane >> 4;
    const _Float16* row = A + (size_t)m * lda;
    v16h a;
#pragma unroll
    for (int v = 0; v < 8; ++v) {
        int k = ((v & 4) << 2) + half * 8 + ((v & 3) << 1);
        a[2 * v]     = row[k];
        a[2 * v + 1] = row[k + 1];
    }
    return a;
}

// B (Kx16 tile) from weights stored row-major (N,K): element(k,n)=W[n][k].
// Lane n reads contiguous K pairs from row n.
__device__ __forceinline__ v16h load_b16(const _Float16* W, int ldw, int n_base,
                                         int k_base, int Nmax, int lane) {
    int n = n_base + (lane & 15);
    if (n >= Nmax) n = Nmax - 1;           // clamp (padded cols never stored)
    int half = lane >> 4;
    const _Float16* row = W + (size_t)n * ldw + k_base;
    v16h b;
#pragma unroll
    for (int v = 0; v < 8; ++v) {
        int k = ((v & 4) << 2) + half * 8 + ((v & 3) << 1);
        b[2 * v]     = row[k];
        b[2 * v + 1] = row[k + 1];
    }
    return b;
}

// B (Kx16) from V stored row-major [row][dim]: element(k,n)=V[row0+k][n],
// rows clamped into [0, LA-1] (out-of-band P entries are zero anyway).
__device__ __forceinline__ v16h load_v16(const _Float16* V, int bb, int base,
                                         int kc, int n_base, int lane) {
    int n = n_base + (lane & 15);
    int half = lane >> 4;
    v16h f;
#pragma unroll
    for (int v = 0; v < 8; ++v) {
        int k  = kc + ((v & 4) << 2) + half * 8 + ((v & 3) << 1);
        int r0 = base + k;     r0 = r0 < 0 ? 0 : (r0 > LA - 1 ? LA - 1 : r0);
        int r1 = base + k + 1; r1 = r1 < 0 ? 0 : (r1 > LA - 1 ? LA - 1 : r1);
        f[2 * v]     = V[((size_t)bb * LA + r0) * 224 + n];
        f[2 * v + 1] = V[((size_t)bb * LA + r1) * 224 + n];
    }
    return f;
}

// ---------------------------------------------------------------------------
// f32 -> f16 convert with K zero-padding (dst is rows x dcols, src rows x scols)
// ---------------------------------------------------------------------------
__global__ void pad_convert_kernel(const float* __restrict__ src,
                                   _Float16* __restrict__ dst,
                                   int rows, int scols, int dcols) {
    size_t total = (size_t)rows * dcols;
    for (size_t i = (size_t)blockIdx.x * blockDim.x + threadIdx.x; i < total;
         i += (size_t)gridDim.x * blockDim.x) {
        int r = (int)(i / dcols), c = (int)(i % dcols);
        dst[i] = (c < scols) ? (_Float16)src[(size_t)r * scols + c] : (_Float16)0.f;
    }
}

__global__ void add_vec_kernel(const float* a, const float* b, float* dst, int n) {
    int i = blockIdx.x * blockDim.x + threadIdx.x;
    if (i < n) dst[i] = a[i] + b[i];
}

// ---------------------------------------------------------------------------
// Generic WMMA GEMM: C[MxN](f32) = A[MxKp](f16) * W[NxKp]^T(f16) + bias, act
// M multiple of 16. act: 0 = none, 1 = ELU.
// ---------------------------------------------------------------------------
__global__ void gemm_wmma_kernel(const _Float16* __restrict__ A, int lda,
                                 const _Float16* __restrict__ W, int ldw,
                                 const float* __restrict__ bias,
                                 float* __restrict__ C, int ldc,
                                 int M, int N, int Kp, int act) {
    const int lane = threadIdx.x & 31;
    const int half = lane >> 4;
    const int col  = lane & 15;
    int gwave  = (blockIdx.x * blockDim.x + threadIdx.x) >> 5;
    int nwaves = (gridDim.x * blockDim.x) >> 5;
    int ntm = M >> 4;
    int ntn = (N + 15) >> 4;
    int total = ntm * ntn;
    for (int tile = gwave; tile < total; tile += nwaves) {
        int tm = tile / ntn, tn = tile % ntn;
        const _Float16* Abase = A + (size_t)(tm * 16) * lda;
        v8f acc = {};
        for (int kc = 0; kc < Kp; kc += 32) {
            __builtin_prefetch(Abase + kc + 64, 0, 3);  // near-cache prefetch
            v16h a = load_a16(Abase + kc, lda, lane);
            v16h b = load_b16(W, ldw, tn * 16, kc, N, lane);
            acc = __builtin_amdgcn_wmma_f32_16x16x32_f16(
                false, a, false, b, (short)0, acc, false, false);
        }
        int n = tn * 16 + col;
        if (n < N) {
            float bv = bias ? bias[n] : 0.f;
#pragma unroll
            for (int r = 0; r < 8; ++r) {
                int m = tm * 16 + r + half * 8;
                float v = acc[r] + bv;
                if (act == 1) v = v > 0.f ? v : (__expf(v) - 1.f);
                C[(size_t)m * ldc + n] = v;
            }
        }
    }
}

// ---------------------------------------------------------------------------
// LSTM sequential scan (one block per direction). w_hh f16 (400x128 padded)
// staged into LDS with CDNA5 async global->LDS DMA; hidden state kept as the
// 16x128 A-fragment (rows 8..15 = 0).
// xg: (B, L, 400) precomputed x@W_ih^T + b_ih + b_hh.
// hout: (B, L, 200) f32, this direction writes cols [out_off, out_off+100).
// ---------------------------------------------------------------------------
__global__ void lstm_scan_kernel(const _Float16* __restrict__ whh,
                                 const float* __restrict__ xg,
                                 float* __restrict__ hout,
                                 int out_off, int reverse) {
    extern __shared__ char smem[];
    _Float16* s_whh = (_Float16*)smem;                  // 400*128 f16
    _Float16* s_h   = s_whh + 400 * 128;                // 16*128  f16
    float*    s_g   = (float*)(s_h + 16 * 128);         // 8*400   f32
    float*    s_c   = s_g + 8 * 400;                    // 8*100   f32

    const int tid  = threadIdx.x;          // 512 threads = 16 waves
    const int lane = tid & 31;
    const int wave = tid >> 5;
    const int half = lane >> 4;
    const int col  = lane & 15;

    {   // 100 KB bulk copy via async DMA (no VGPR round-trip), 16B per lane
        const char* g = (const char*)whh;
        char*       l = (char*)s_whh;
        for (int i = tid * 16; i < 400 * 128 * 2; i += blockDim.x * 16)
            async_copy_b128(l + i, g + i);
        wait_asynccnt0();
    }
    for (int i = tid; i < 16 * 128;  i += blockDim.x) s_h[i] = (_Float16)0.f;
    for (int i = tid; i < 800;       i += blockDim.x) s_c[i] = 0.f;
    __syncthreads();

    for (int t = 0; t < LA; ++t) {
        int tt = reverse ? (LA - 1 - t) : t;
        // g(8x400) = h(8x100) @ whh^T  (25 N-tiles over 16 waves)
#pragma unroll
        for (int rep = 0; rep < 2; ++rep) {
            int tile = wave + rep * 16;
            if (tile < 25) {
                v8f acc = {};
#pragma unroll
                for (int kc = 0; kc < 4; ++kc) {
                    v16h a = load_a16(s_h + kc * 32, 128, lane);
                    v16h b = load_b16(s_whh, 128, tile * 16, kc * 32, 400, lane);
                    acc = __builtin_amdgcn_wmma_f32_16x16x32_f16(
                        false, a, false, b, (short)0, acc, false, false);
                }
                int n = tile * 16 + col;
                if (half == 0) {   // rows 0..7 are valid batch rows
#pragma unroll
                    for (int r = 0; r < 8; ++r)
                        s_g[r * 400 + n] =
                            acc[r] + xg[((size_t)r * LA + tt) * 400 + n];
                }
            }
        }
        __syncthreads();
        // pointwise gates (torch order i,f,g,o)
        if (tid < 800) {
            int m = tid / 100, ix = tid % 100;
            const float* gm = s_g + m * 400;
            float si = 1.f / (1.f + __expf(-gm[ix]));
            float sf = 1.f / (1.f + __expf(-gm[100 + ix]));
            float tc = tanhf(gm[200 + ix]);
            float so = 1.f / (1.f + __expf(-gm[300 + ix]));
            float c  = sf * s_c[tid] + si * tc;
            s_c[tid] = c;
            float h  = so * tanhf(c);
            s_h[m * 128 + ix] = (_Float16)h;
            hout[((size_t)m * LA + tt) * 200 + out_off + ix] = h;
        }
        __syncthreads();
    }
}

// ---------------------------------------------------------------------------
// Banded attention, one block per (batch, 16-row tile).
// Full-row QK^T for the softmax denominator (reference normalizes over all L,
// THEN masks), band tiles (<=9) kept as masked exp() in LDS, then P*V via WMMA.
// ---------------------------------------------------------------------------
__global__ void attention_kernel(const _Float16* __restrict__ Q,
                                 const _Float16* __restrict__ K,
                                 const _Float16* __restrict__ V,
                                 float* __restrict__ ctx) {
    __shared__ _Float16 P[16 * 160];
    __shared__ float rowsum[16];

    const int blk = blockIdx.x;
    const int bb  = blk >> 7;        // batch
    const int rt  = blk & 127;       // row tile
    const int tid  = threadIdx.x;    // 256 = 8 waves
    const int lane = tid & 31;
    const int wave = tid >> 5;
    const int half = lane >> 4;
    const int col  = lane & 15;

    for (int i = tid; i < 16 * 160; i += blockDim.x) P[i] = (_Float16)0.f;
    if (tid < 16) rowsum[tid] = 0.f;
    __syncthreads();

    const _Float16* Qrow = Q + ((size_t)bb * LA + rt * 16) * 224;
    v16h qf[7];
#pragma unroll
    for (int kc = 0; kc < 7; ++kc) qf[kc] = load_a16(Qrow + kc * 32, 224, lane);

    float psum[8];
#pragma unroll
    for (int r = 0; r < 8; ++r) psum[r] = 0.f;
    const int bt0 = rt - 4;

    // Phase 1: scores over all 128 column tiles
    for (int mt = wave; mt < 128; mt += 8) {
        v8f acc = {};
#pragma unroll
        for (int kc = 0; kc < 7; ++kc) {
            v16h kf = load_b16(K + (size_t)bb * LA * 224, 224, mt * 16,
                               kc * 32, LA, lane);
            acc = __builtin_amdgcn_wmma_f32_16x16x32_f16(
                false, qf[kc], false, kf, (short)0, acc, false, false);
        }
        float e[8];
#pragma unroll
        for (int r = 0; r < 8; ++r) {
            e[r] = __expf(acc[r] * (1.f / 256.f));   // hardcoded /256 scale
            psum[r] += e[r];
        }
        if (mt >= bt0 && mt <= rt + 4) {
            int lt = mt - bt0;
            int j  = mt * 16 + col;
#pragma unroll
            for (int r = 0; r < 8; ++r) {
                int ml = r + half * 8;
                int i  = rt * 16 + ml;
                int d  = i - j; if (d < 0) d = -d;
                P[ml * 160 + lt * 16 + col] =
                    (d <= 64) ? (_Float16)e[r] : (_Float16)0.f;
            }
        }
    }
#pragma unroll
    for (int r = 0; r < 8; ++r) atomicAdd(&rowsum[r + half * 8], psum[r]);
    __syncthreads();

    // Phase 2: context = P(16x160) * V(160x200), A-fragments from LDS
    for (int tn = wave; tn < 13; tn += 8) {
        v8f acc = {};
#pragma unroll
        for (int kc = 0; kc < 5; ++kc) {
            v16h a = load_a16(P + kc * 32, 160, lane);
            v16h b = load_v16(V, bb, bt0 * 16, kc * 32, tn * 16, lane);
            acc = __builtin_amdgcn_wmma_f32_16x16x32_f16(
                false, a, false, b, (short)0, acc, false, false);
        }
        int n = tn * 16 + col;
        if (n < 200) {
#pragma unroll
            for (int r = 0; r < 8; ++r) {
                int ml = r + half * 8;
                int i  = rt * 16 + ml;
                ctx[((size_t)bb * LA + i) * 200 + n] = acc[r] / rowsum[ml];
            }
        }
    }
}

// ---------------------------------------------------------------------------
// frame_score[b,l] = l1[b,l,:] . l2_w + l2_b
// ---------------------------------------------------------------------------
__global__ void frame_kernel(const float* __restrict__ l1,
                             const float* __restrict__ l2w,
                             const float* __restrict__ l2b,
                             float* __restrict__ out) {
    for (int i = blockIdx.x * blockDim.x + threadIdx.x; i < BB * LA;
         i += gridDim.x * blockDim.x) {
        float s = l2b[0];
        const float* row = l1 + (size_t)i * 50;
#pragma unroll 10
        for (int j = 0; j < 50; ++j) s += row[j] * l2w[j];
        out[i] = s;
    }
}

__global__ void mean_kernel(const float* __restrict__ frame,
                            float* __restrict__ avg) {
    __shared__ float red[256];
    int b = blockIdx.x;
    float s = 0.f;
    for (int i = threadIdx.x; i < LA; i += blockDim.x)
        s += frame[(size_t)b * LA + i];
    red[threadIdx.x] = s;
    __syncthreads();
    for (int st = 128; st > 0; st >>= 1) {
        if (threadIdx.x < st) red[threadIdx.x] += red[threadIdx.x + st];
        __syncthreads();
    }
    if (threadIdx.x == 0) avg[b] = red[0] * (1.f / LA);
}

// ---------------------------------------------------------------------------
extern "C" void kernel_launch(void* const* d_in, const int* in_sizes, int n_in,
                              void* d_out, int out_size, void* d_ws, size_t ws_size,
                              hipStream_t stream) {
    (void)in_sizes; (void)n_in; (void)out_size; (void)ws_size;
    const float* x        = (const float*)d_in[0];
    const float* wih[4]   = {(const float*)d_in[1], (const float*)d_in[5],
                             (const float*)d_in[9], (const float*)d_in[13]};
    const float* whh[4]   = {(const float*)d_in[2], (const float*)d_in[6],
                             (const float*)d_in[10], (const float*)d_in[14]};
    const float* bih[4]   = {(const float*)d_in[3], (const float*)d_in[7],
                             (const float*)d_in[11], (const float*)d_in[15]};
    const float* bhh[4]   = {(const float*)d_in[4], (const float*)d_in[8],
                             (const float*)d_in[12], (const float*)d_in[16]};
    const float* wq_w = (const float*)d_in[17]; const float* wq_b = (const float*)d_in[18];
    const float* wk_w = (const float*)d_in[19]; const float* wk_b = (const float*)d_in[20];
    const float* wv_w = (const float*)d_in[21]; const float* wv_b = (const float*)d_in[22];
    const float* l1_w = (const float*)d_in[23]; const float* l1_b = (const float*)d_in[24];
    const float* l2_w = (const float*)d_in[25]; const float* l2_b = (const float*)d_in[26];
    float* out = (float*)d_out;

    // ---- workspace bump allocator ----
    char* ws = (char*)d_ws;
    size_t off = 0;
    auto alloc = [&](size_t bytes) -> char* {
        char* p = ws + off;
        off += (bytes + 255) & ~(size_t)255;
        return p;
    };
    const int ML = BB * LA;  // 16384 rows
    _Float16* x16     = (_Float16*)alloc((size_t)ML * 288 * 2);
    _Float16* wih016[2] = {(_Float16*)alloc(400 * 288 * 2), (_Float16*)alloc(400 * 288 * 2)};
    _Float16* wih116[2] = {(_Float16*)alloc(400 * 224 * 2), (_Float16*)alloc(400 * 224 * 2)};
    _Float16* whh16[4];
    for (int i = 0; i < 4; ++i) whh16[i] = (_Float16*)alloc(400 * 128 * 2);
    _Float16* wq16 = (_Float16*)alloc(200 * 224 * 2);
    _Float16* wk16 = (_Float16*)alloc(200 * 224 * 2);
    _Float16* wv16 = (_Float16*)alloc(200 * 224 * 2);
    _Float16* l1w16 = (_Float16*)alloc(50 * 224 * 2);
    float* bsum[4];
    for (int i = 0; i < 4; ++i) bsum[i] = (float*)alloc(400 * 4);
    float* xgf = (float*)alloc((size_t)ML * 400 * 4);   // reused layer0 -> layer1
    float* xgb = (float*)alloc((size_t)ML * 400 * 4);
    float* hbuf = (float*)alloc((size_t)ML * 200 * 4);  // reused l0 out -> l1 out
    _Float16* h016   = (_Float16*)alloc((size_t)ML * 224 * 2);
    _Float16* lstm16 = (_Float16*)alloc((size_t)ML * 224 * 2);
    float* qf = (float*)alloc((size_t)ML * 200 * 4);
    float* kf = (float*)alloc((size_t)ML * 200 * 4);
    float* vf = (float*)alloc((size_t)ML * 200 * 4);
    _Float16* q16 = (_Float16*)alloc((size_t)ML * 224 * 2);
    _Float16* k16 = (_Float16*)alloc((size_t)ML * 224 * 2);
    _Float16* v16 = (_Float16*)alloc((size_t)ML * 224 * 2);
    float* ctxf = (float*)alloc((size_t)ML * 200 * 4);
    _Float16* ctx16 = (_Float16*)alloc((size_t)ML * 224 * 2);
    float* l1f = (float*)alloc((size_t)ML * 50 * 4);

    const size_t scan_smem = 400 * 128 * 2 + 16 * 128 * 2 + 8 * 400 * 4 + 800 * 4;

    // ---- stage 0: conversions ----
    pad_convert_kernel<<<2048, 256, 0, stream>>>(x, x16, ML, 257, 288);
    pad_convert_kernel<<<256, 256, 0, stream>>>(wih[0], wih016[0], 400, 257, 288);
    pad_convert_kernel<<<256, 256, 0, stream>>>(wih[1], wih016[1], 400, 257, 288);
    pad_convert_kernel<<<256, 256, 0, stream>>>(wih[2], wih116[0], 400, 200, 224);
    pad_convert_kernel<<<256, 256, 0, stream>>>(wih[3], wih116[1], 400, 200, 224);
    for (int i = 0; i < 4; ++i) {
        pad_convert_kernel<<<128, 256, 0, stream>>>(whh[i], whh16[i], 400, 100, 128);
        add_vec_kernel<<<2, 256, 0, stream>>>(bih[i], bhh[i], bsum[i], 400);
    }
    pad_convert_kernel<<<128, 256, 0, stream>>>(wq_w, wq16, 200, 200, 224);
    pad_convert_kernel<<<128, 256, 0, stream>>>(wk_w, wk16, 200, 200, 224);
    pad_convert_kernel<<<128, 256, 0, stream>>>(wv_w, wv16, 200, 200, 224);
    pad_convert_kernel<<<64, 256, 0, stream>>>(l1_w, l1w16, 50, 200, 224);

    // ---- layer 0: gate GEMMs + scans ----
    gemm_wmma_kernel<<<512, 256, 0, stream>>>(x16, 288, wih016[0], 288, bsum[0],
                                              xgf, 400, ML, 400, 288, 0);
    gemm_wmma_kernel<<<512, 256, 0, stream>>>(x16, 288, wih016[1], 288, bsum[1],
                                              xgb, 400, ML, 400, 288, 0);
    lstm_scan_kernel<<<1, 512, scan_smem, stream>>>(whh16[0], xgf, hbuf, 0, 0);
    lstm_scan_kernel<<<1, 512, scan_smem, stream>>>(whh16[1], xgb, hbuf, 100, 1);
    pad_convert_kernel<<<2048, 256, 0, stream>>>(hbuf, h016, ML, 200, 224);

    // ---- layer 1 ----
    gemm_wmma_kernel<<<512, 256, 0, stream>>>(h016, 224, wih116[0], 224, bsum[2],
                                              xgf, 400, ML, 400, 224, 0);
    gemm_wmma_kernel<<<512, 256, 0, stream>>>(h016, 224, wih116[1], 224, bsum[3],
                                              xgb, 400, ML, 400, 224, 0);
    lstm_scan_kernel<<<1, 512, scan_smem, stream>>>(whh16[2], xgf, hbuf, 0, 0);
    lstm_scan_kernel<<<1, 512, scan_smem, stream>>>(whh16[3], xgb, hbuf, 100, 1);
    pad_convert_kernel<<<2048, 256, 0, stream>>>(hbuf, lstm16, ML, 200, 224);

    // ---- attention projections ----
    gemm_wmma_kernel<<<512, 256, 0, stream>>>(lstm16, 224, wq16, 224, wq_b,
                                              qf, 200, ML, 200, 224, 0);
    gemm_wmma_kernel<<<512, 256, 0, stream>>>(lstm16, 224, wk16, 224, wk_b,
                                              kf, 200, ML, 200, 224, 0);
    gemm_wmma_kernel<<<512, 256, 0, stream>>>(lstm16, 224, wv16, 224, wv_b,
                                              vf, 200, ML, 200, 224, 0);
    pad_convert_kernel<<<2048, 256, 0, stream>>>(qf, q16, ML, 200, 224);
    pad_convert_kernel<<<2048, 256, 0, stream>>>(kf, k16, ML, 200, 224);
    pad_convert_kernel<<<2048, 256, 0, stream>>>(vf, v16, ML, 200, 224);

    // ---- banded attention ----
    attention_kernel<<<BB * (LA / 16), 256, 0, stream>>>(q16, k16, v16, ctxf);
    pad_convert_kernel<<<2048, 256, 0, stream>>>(ctxf, ctx16, ML, 200, 224);

    // ---- MLP head ----
    gemm_wmma_kernel<<<512, 256, 0, stream>>>(ctx16, 224, l1w16, 224, l1_b,
                                              l1f, 50, ML, 50, 224, 1 /*ELU*/);
    frame_kernel<<<128, 256, 0, stream>>>(l1f, l2_w, l2_b, out);
    mean_kernel<<<BB, 256, 0, stream>>>(out, out + ML);
}
